// OverFitter_81578608820661
// MI455X (gfx1250) — matrix operand order repaired and checked by reading, not compiled, not run on hardware
//
#include <hip/hip_runtime.h>
#include <math.h>

#define HH 1024
#define WW 1536
#define HWN (HH * WW)
#define IMG_SIZE_F ((float)(HWN * 3))

typedef __attribute__((ext_vector_type(16))) _Float16 v16h;
typedef __attribute__((ext_vector_type(8)))  float    v8f;

// ---------------- WMMA helpers (gfx1250 16x16x32 f16 -> f32) ----------------

__device__ __forceinline__ v16h h16z() { v16h v = {}; return v; }

__device__ __forceinline__ v8f wmma32(v16h A, v16h B, v8f C) {
  // (neg_a, A, neg_b, B, c_mod, C, reuse_a, reuse_b)
  return __builtin_amdgcn_wmma_f32_16x16x32_f16(false, A, false, B, (short)0, C, false, false);
}

// A-matrix (16x32 f16): lane l holds row m=l&15; slots j=0..15 -> K=base+j,
// base = (l>=16)?16:0.
// Weights stored K-major: W[k*M + m]  (arm_w1/w2/w3)
__device__ __forceinline__ v16h build_A_kmajor(const float* Wm, int K, int M, int lane) {
  int m = lane & 15; int base = (lane & 16) ? 16 : 0;
  v16h a = h16z();
#pragma unroll
  for (int j = 0; j < 16; ++j) {
    int k = base + j;
    float v = (k < K && m < M) ? Wm[k * M + m] : 0.0f;
    a[j] = (_Float16)v;
  }
  return a;
}

// Weights stored M-major: W[m*Ktot + k]  (syn_w1 (40,7), syn_w2 (9,40))
__device__ __forceinline__ v16h build_A_mmajor(const float* Wm, int Ktot, int Mtot,
                                               int kofs, int mofs, int lane) {
  int m = mofs + (lane & 15); int base = (lane & 16) ? 16 : 0;
  v16h a = h16z();
#pragma unroll
  for (int j = 0; j < 16; ++j) {
    int kl = base + j;
    int k = kofs + kl;
    float v = (k < Ktot && m < Mtot) ? Wm[m * Ktot + k] : 0.0f;
    a[j] = (_Float16)v;
  }
  return a;
}

// C/D f32 tile: lane l, VGPR r -> (M = r + ((l>=16)?8:0), N = l&15). Bias broadcast.
__device__ __forceinline__ v8f build_C_bias(const float* b, int n, int ofs, int lane) {
  int base = (lane & 16) ? 8 : 0;
  v8f c = {};
#pragma unroll
  for (int r = 0; r < 8; ++r) {
    int m = ofs + base + r;
    c[r] = (m < n) ? b[m] : 0.0f;
  }
  return c;
}

// B-matrix (32x16 f16): lane l holds col n=l&15; lanes 0-15 slots j -> K=j,
// lanes 16-31 slots j -> K=16+j.
// Repack one f32 D tile (hidden 0..15) into B with relu; K=16..31 zero.
__device__ __forceinline__ v16h pack_B_single(v8f c, bool lo) {
  float p[8];
#pragma unroll
  for (int r = 0; r < 8; ++r) p[r] = __shfl_xor(c[r], 16, 32);
  v16h b = h16z();
#pragma unroll
  for (int j = 0; j < 16; ++j) {
    float v = lo ? ((j < 8) ? c[j] : p[j - 8]) : 0.0f;
    b[j] = (_Float16)fmaxf(v, 0.0f);
  }
  return b;
}

// Repack two f32 D tiles (hidden 0..15 in c0, 16..31 in c1) into one B (K=0..31), relu.
__device__ __forceinline__ v16h pack_B_pair(v8f c0, v8f c1, bool lo) {
  float p0[8], p1[8];
#pragma unroll
  for (int r = 0; r < 8; ++r) { p0[r] = __shfl_xor(c0[r], 16, 32); p1[r] = __shfl_xor(c1[r], 16, 32); }
  v16h b = h16z();
#pragma unroll
  for (int j = 0; j < 16; ++j) {
    float v = lo ? ((j < 8) ? c0[j] : p0[j - 8])
                 : ((j < 8) ? p1[j] : c1[j - 8]);
    b[j] = (_Float16)fmaxf(v, 0.0f);
  }
  return b;
}

__device__ __forceinline__ float sigmf(float z) { return 1.0f / (1.0f + expf(-z)); }

__device__ __forceinline__ float lap_cdf(float v, float b) {
  float s = (v >= 0.0f) ? 1.0f : -1.0f;
  return 0.5f - 0.5f * s * expm1f(-fabsf(v) / b);
}

__device__ __forceinline__ void block_accum(float v, float* gacc) {
#pragma unroll
  for (int o = 16; o > 0; o >>= 1) v += __shfl_down(v, o, 32);
  __shared__ float red[8];
  int lane = threadIdx.x & 31, wid = threadIdx.x >> 5;
  if (lane == 0) red[wid] = v;
  __syncthreads();
  if (threadIdx.x == 0) {
    float s = 0.0f;
#pragma unroll
    for (int i = 0; i < 8; ++i) s += red[i];
    atomicAdd(gacc, s);
  }
}

// ---------------- elementwise kernels ----------------

__global__ void k_init(float* a) { a[0] = 0.0f; a[1] = 0.0f; }

__global__ void k_quant(const float* in, _Float16* out, int n) {
  int i = blockIdx.x * blockDim.x + threadIdx.x;
  if (i < n) {
    float v = rintf(in[i] * 64.0f);           // GAIN=64, round-to-nearest-even
    v = fminf(fmaxf(v, -31.0f), 31.0f);       // LATENT_MAXV
    out[i] = (_Float16)v;
  }
}

// conv_transpose stride2 k=8 SAME: even o -> taps K[0,2,4,6] at i=o/2-2+j;
// odd o -> taps K[1,3,5,7] at i=(o-3)/2+j. 2D kernel = K8[kr*8+kc].
__global__ void k_upsample(const _Float16* __restrict__ in, _Float16* __restrict__ out,
                           int hin, int win, const float* __restrict__ K8) {
  int idx = blockIdx.x * blockDim.x + threadIdx.x;
  int wout = win * 2;
  int tot = hin * 2 * wout;
  if (idx >= tot) return;
  int oy = idx / wout, ox = idx - oy * wout;
  int ry = oy & 1, rx = ox & 1;
  int by = (oy >> 1) - (ry ? 1 : 2);
  int bx = (ox >> 1) - (rx ? 1 : 2);
  float acc = 0.0f;
#pragma unroll
  for (int a = 0; a < 4; ++a) {
    int iy = by + a;
    if (iy < 0 || iy >= hin) continue;
    int kr = 2 * a + ry;
#pragma unroll
    for (int b = 0; b < 4; ++b) {
      int ix = bx + b;
      if (ix < 0 || ix >= win) continue;
      int kc = 2 * b + rx;
      acc += K8[kr * 8 + kc] * (float)in[iy * win + ix];
    }
  }
  out[idx] = (_Float16)acc;
}

// ---------------- ARM entropy MLP (2->16->16->2) + Laplace rate ----------------
// 32 pixels per wave iteration: two 16-pixel WMMA column tiles; all loads and the
// Laplace/transcendental tail run full-wave and branchless.

__global__ __launch_bounds__(256) void k_arm(const _Float16* __restrict__ q, int h, int w,
    const float* aw1, const float* ab1, const float* aw2, const float* ab2,
    const float* aw3, const float* ab3, float* gacc) {
  int lane = threadIdx.x & 31;
  int wave = (blockIdx.x * blockDim.x + threadIdx.x) >> 5;
  int nwaves = (gridDim.x * blockDim.x) >> 5;
  int npix = h * w;
  int ntiles = npix >> 5;              // all latent sizes divisible by 32
  bool lo = (lane & 16) == 0;

  v16h A1 = build_A_kmajor(aw1, 2, 16, lane);   // arm_w1 (2,16)
  v16h A2 = build_A_kmajor(aw2, 16, 16, lane);  // arm_w2 (16,16)
  v16h A3 = build_A_kmajor(aw3, 16, 2, lane);   // arm_w3 (16,2)
  v8f Cb1 = build_C_bias(ab1, 16, 0, lane);
  v8f Cb2 = build_C_bias(ab2, 16, 0, lane);
  v8f Cb3 = build_C_bias(ab3, 2, 0, lane);

  float racc = 0.0f;
  for (int t = wave; t < ntiles; t += nwaves) {
    int p = (t << 5) + lane;           // one pixel per lane, coalesced
    int y = p / w, x = p - y * w;
    float lat = (float)q[p];
    // branchless context: clamp address, then select zero at the border
    float upv = (float)q[(y > 0) ? (p - w) : p];
    float lfv = (float)q[(x > 0) ? (p - 1) : p];
    float up = (y > 0) ? upv : 0.0f;
    float lf = (x > 0) ? lfv : 0.0f;
    float upx = __shfl_xor(up, 16, 32);
    float lfx = __shfl_xor(lf, 16, 32);

    v16h B1a = h16z();                 // tile0: pixels base..base+15
    v16h B1b = h16z();                 // tile1: pixels base+16..base+31
    B1a[0] = (_Float16)(lo ? up  : 0.0f);
    B1a[1] = (_Float16)(lo ? lf  : 0.0f);
    B1b[0] = (_Float16)(lo ? upx : 0.0f);
    B1b[1] = (_Float16)(lo ? lfx : 0.0f);

    v8f ca = wmma32(A1, B1a, Cb1);
    v8f cb = wmma32(A1, B1b, Cb1);
    v16h B2a = pack_B_single(ca, lo);
    v16h B2b = pack_B_single(cb, lo);
    ca = wmma32(A2, B2a, Cb2);
    cb = wmma32(A2, B2b, Cb2);
    v16h B3a = pack_B_single(ca, lo);
    v16h B3b = pack_B_single(cb, lo);
    ca = wmma32(A3, B3a, Cb3);
    cb = wmma32(A3, B3b, Cb3);

    // transpose to lane-per-pixel: lanes<16 own tile0 cols, lanes>=16 take tile1 cols
    float mux = __shfl_xor(cb[0], 16, 32);
    float lbx = __shfl_xor(cb[1], 16, 32);
    float mu   = lo ? ca[0] : mux;
    float logb = lo ? ca[1] : lbx;

    float bb = fminf(fmaxf(expf(logb), 1e-3f), 150.0f);
    float pr = lap_cdf(lat + 0.5f - mu, bb) - lap_cdf(lat - 0.5f - mu, bb);
    pr = fminf(fmaxf(pr, 0.000244140625f), 1.0f);   // 2^-12
    racc -= log2f(pr);
  }
  block_accum(racc, gacc);
}

// ---------------- synthesis MLP (7->40->9) + prior + image rate ----------------

struct SynPtrs { const _Float16* ch[7]; };

__global__ __launch_bounds__(256) void k_syn(SynPtrs chs,
    const float* __restrict__ img, const float* __restrict__ prior,
    const float* __restrict__ coeffs,
    const float* sw1, const float* sb1, const float* sw2, const float* sb2,
    float* gacc) {
  int lane = threadIdx.x & 31;
  int wave = (blockIdx.x * blockDim.x + threadIdx.x) >> 5;
  int nwaves = (gridDim.x * blockDim.x) >> 5;
  const int ntiles = HWN / 32;
  bool lo = (lane & 16) == 0;

  v16h A10 = build_A_mmajor(sw1, 7, 40, 0, 0, lane);   // hidden 0-15
  v16h A11 = build_A_mmajor(sw1, 7, 40, 0, 16, lane);  // hidden 16-31
  v16h A12 = build_A_mmajor(sw1, 7, 40, 0, 32, lane);  // hidden 32-47 (40-47 pad)
  v16h A2a = build_A_mmajor(sw2, 40, 9, 0, 0, lane);   // K 0-31
  v16h A2b = build_A_mmajor(sw2, 40, 9, 32, 0, lane);  // K 32-63 (40-63 pad)
  v8f Cb10 = build_C_bias(sb1, 40, 0, lane);
  v8f Cb11 = build_C_bias(sb1, 40, 16, lane);
  v8f Cb12 = build_C_bias(sb1, 40, 32, lane);
  v8f Cb2  = build_C_bias(sb2, 9, 0, lane);

  float racc = 0.0f;
  for (int t = wave; t < ntiles; t += nwaves) {
    int p = (t << 5) + lane;           // one pixel per lane, coalesced

    // all 32 lanes load their own pixel's 7 channels
    float chv[7];
#pragma unroll
    for (int j = 0; j < 7; ++j) chv[j] = (float)chs.ch[j][p];
    float chx[7];
#pragma unroll
    for (int j = 0; j < 7; ++j) chx[j] = __shfl_xor(chv[j], 16, 32);

    v16h B1a = h16z();                 // tile0: pixels base..base+15
    v16h B1b = h16z();                 // tile1: pixels base+16..base+31
#pragma unroll
    for (int j = 0; j < 7; ++j) {
      B1a[j] = (_Float16)(lo ? chv[j] : 0.0f);
      B1b[j] = (_Float16)(lo ? chx[j] : 0.0f);
    }

    // tile0 MLP
    v8f c0 = wmma32(A10, B1a, Cb10);
    v8f c1 = wmma32(A11, B1a, Cb11);
    v8f c2 = wmma32(A12, B1a, Cb12);
    v16h B2p = pack_B_pair(c0, c1, lo);
    v16h B2s = pack_B_single(c2, lo);
    v8f da = wmma32(A2a, B2p, Cb2);
    da = wmma32(A2b, B2s, da);
    // tile1 MLP
    c0 = wmma32(A10, B1b, Cb10);
    c1 = wmma32(A11, B1b, Cb11);
    c2 = wmma32(A12, B1b, Cb12);
    B2p = pack_B_pair(c0, c1, lo);
    B2s = pack_B_single(c2, lo);
    v8f db = wmma32(A2a, B2p, Cb2);
    db = wmma32(A2b, B2s, db);

    // transpose to lane-per-pixel: e[0..8] = all 9 synthesis params for this lane's pixel
    float e[9];
    {
      float dbx[8];
#pragma unroll
      for (int r = 0; r < 8; ++r) dbx[r] = __shfl_xor(db[r], 16, 32);
      float da0x = __shfl_xor(da[0], 16, 32);
#pragma unroll
      for (int r = 0; r < 8; ++r) e[r] = lo ? da[r] : dbx[r];
      e[8] = lo ? da0x : db[0];
    }

    // prior with parity coefficients (full-wave, coalesced)
    int y = p / WW, x = p - y * WW;
    int par = ((y & 1) << 1) | (x & 1);
#pragma unroll
    for (int r = 0; r < 9; ++r)
      e[r] += coeffs[par * 9 + r] * prior[(size_t)r * HWN + p];

    float xr = img[p], xg = img[HWN + p], xb = img[2 * HWN + p];
    float mu[3];
    mu[0] = e[0];
    mu[1] = e[1] + e[6] * xr;
    mu[2] = e[2] + e[7] * xr + e[8] * xg;
    float xs[3] = {xr, xg, xb};
    const float hq = 0.5f / 255.0f;
#pragma unroll
    for (int cch = 0; cch < 3; ++cch) {
      float s = fminf(fmaxf(expf(e[3 + cch]), 1e-4f), 10.0f);
      float a = sigmf((xs[cch] + hq - mu[cch]) / s);
      float b = sigmf((xs[cch] - hq - mu[cch]) / s);
      float pi = fminf(fmaxf(a - b, 3.0517578125e-5f), 1.0f);  // 2^-15
      racc -= log2f(pi);
    }
  }
  block_accum(racc, gacc);
}

__global__ void k_final(const float* a, float* out) {
  float lb = a[0] / IMG_SIZE_F;
  float ib = a[1] / IMG_SIZE_F;
  out[0] = lb; out[1] = ib; out[2] = lb + ib;
}

// ---------------- host orchestration ----------------

extern "C" void kernel_launch(void* const* d_in, const int* in_sizes, int n_in,
                              void* d_out, int out_size, void* d_ws, size_t ws_size,
                              hipStream_t stream) {
  static const int QS[7] = {1572864, 393216, 98304, 24576, 6144, 1536, 384};
  const float* img   = (const float*)d_in[0];
  const float* prior = (const float*)d_in[1];
  const float* lat[7];
  int wi;
  if (n_in >= 21) {                      // latents flattened into 7 inputs
    for (int i = 0; i < 7; ++i) lat[i] = (const float*)d_in[2 + i];
    wi = 9;
  } else {                               // latents concatenated into one buffer
    const float* base = (const float*)d_in[2];
    size_t o = 0;
    for (int i = 0; i < 7; ++i) { lat[i] = base + o; o += (size_t)QS[i]; }
    wi = 3;
  }
  const float* coeffs = (const float*)d_in[wi + 0];
  const float* aw1 = (const float*)d_in[wi + 1];
  const float* ab1 = (const float*)d_in[wi + 2];
  const float* aw2 = (const float*)d_in[wi + 3];
  const float* ab2 = (const float*)d_in[wi + 4];
  const float* aw3 = (const float*)d_in[wi + 5];
  const float* ab3 = (const float*)d_in[wi + 6];
  const float* sw1 = (const float*)d_in[wi + 7];
  const float* sb1 = (const float*)d_in[wi + 8];
  const float* sw2 = (const float*)d_in[wi + 9];
  const float* sb2 = (const float*)d_in[wi + 10];
  const float* upsk = (const float*)d_in[wi + 11];

  _Float16* hw = (_Float16*)d_ws;
  size_t qo[7]; { size_t o = 0; for (int i = 0; i < 7; ++i) { qo[i] = o; o += (size_t)QS[i]; } }
  size_t lpbase = qo[6] + (size_t)QS[6];            // 2,097,024 halves
  size_t tmpA = lpbase + (size_t)6 * HWN;
  size_t tmpB = tmpA + 393216;
  float* accs = (float*)((char*)d_ws + (tmpB + 393216) * sizeof(_Float16));

  k_init<<<1, 1, 0, stream>>>(accs);

  for (int i = 0; i < 7; ++i)
    k_quant<<<(QS[i] + 255) / 256, 256, 0, stream>>>(lat[i], hw + qo[i], QS[i]);

  // upsample cascades: latent i needs i doublings; final stage lands in lp channel
  for (int i = 1; i < 7; ++i) {
    const _Float16* src = hw + qo[i];
    int hs = HH >> i, wsz = WW >> i;
    for (int s = i; s >= 1; --s) {
      _Float16* dst = (s == 1) ? (hw + lpbase + (size_t)(i - 1) * HWN)
                               : (hw + ((((i - s) & 1) == 0) ? tmpA : tmpB));
      int outN = (hs * 2) * (wsz * 2);
      k_upsample<<<(outN + 255) / 256, 256, 0, stream>>>(src, dst, hs, wsz, upsk);
      src = dst; hs <<= 1; wsz <<= 1;
    }
  }

  for (int i = 0; i < 7; ++i) {
    int ntiles = QS[i] / 32;
    int blocks = (ntiles + 7) / 8;
    if (blocks > 6144) blocks = 6144;
    if (blocks < 1) blocks = 1;
    k_arm<<<blocks, 256, 0, stream>>>(hw + qo[i], HH >> i, WW >> i,
                                      aw1, ab1, aw2, ab2, aw3, ab3, accs);
  }

  SynPtrs sp;
  sp.ch[0] = hw + qo[0];
  for (int i = 1; i < 7; ++i) sp.ch[i] = hw + lpbase + (size_t)(i - 1) * HWN;
  k_syn<<<4096, 256, 0, stream>>>(sp, img, prior, coeffs, sw1, sb1, sw2, sb2, accs + 1);

  k_final<<<1, 1, 0, stream>>>(accs, (float*)d_out);

  (void)in_sizes; (void)out_size; (void)ws_size;
}
